// GCN_86466281603780
// MI455X (gfx1250) — matrix operand order repaired and checked by reading, not compiled, not run on hardware
//
#include <hip/hip_runtime.h>
#include <hip/hip_bf16.h>

// ---------------------------------------------------------------------------
// GCN forward for MI455X (gfx1250, wave32).
//   x:[50000,128] -> GCNConv(W1)->BN->ReLU -> GCNConv(W2)->BN->ReLU
//     -> GCNConv(W3) -> log_softmax
// GEMMs use v_wmma_f32_16x16x32_bf16 (bf16 inputs, f32 accumulate), with the
// C tile staged through LDS for fully coalesced 128B stores.
// Message passing uses f32 global atomics (L2-resident: h matrix is 51MB,
// MI455X L2 is 192MB).
// ---------------------------------------------------------------------------

#define NNODES 50000
#define NEDGES 800000
#define IN_DIM 128
#define HIDD   256
#define OUTD   40
#define OUTP   48          // OUTD padded to 16
#define BN_EPS 1e-5f

typedef __attribute__((ext_vector_type(16))) __bf16 v16bf;
typedef __attribute__((ext_vector_type(8)))  float  v8f;

__device__ __forceinline__ __bf16 f2bf(float f) {
  unsigned u = __builtin_bit_cast(unsigned, f);
  u += 0x7FFFu + ((u >> 16) & 1u);             // round-to-nearest-even
  unsigned short h = (unsigned short)(u >> 16);
  return __builtin_bit_cast(__bf16, h);
}

// ---------------------------------------------------------------- utilities
__global__ void fill_f32(float* p, int n, float v) {
  int t = blockIdx.x * blockDim.x + threadIdx.x;
  if (t < n) p[t] = v;
}

__global__ void rsqrt_inplace(float* p, int n) {
  int t = blockIdx.x * blockDim.x + threadIdx.x;
  if (t < n) p[t] = rsqrtf(p[t]);
}

// -------------------------------------------------------------- graph norm
__global__ void degree_accum(const int* __restrict__ col, float* __restrict__ deg) {
  int e = blockIdx.x * blockDim.x + threadIdx.x;
  if (e < NEDGES) unsafeAtomicAdd(&deg[col[e]], 1.0f);
}

__global__ void edge_norm(const int* __restrict__ row, const int* __restrict__ col,
                          const float* __restrict__ dinv, float* __restrict__ norm) {
  int e = blockIdx.x * blockDim.x + threadIdx.x;
  if (e < NEDGES) norm[e] = dinv[row[e]] * dinv[col[e]];
}

// -------------------------------------------------- weight packing (f32->bf16)
// Layout: for K-step s (32 wide), 16-col tile, lane 0..31: the 16 bf16 values
// of that lane's B fragment stored contiguously (one 32B load in the GEMM).
// Lane l covers col n = tile*16 + (l&15), k = s*32 + (l>>4)*8 + (e<8 ? e : e+8), e=0..15.
__global__ void pack_weight_bf16(const float* __restrict__ W, __bf16* __restrict__ Wp,
                                 int K, int Nvalid, int nColTiles) {
  int tid = blockIdx.x * blockDim.x + threadIdx.x;
  int total = (K >> 5) * nColTiles * 32;
  if (tid >= total) return;
  int lane = tid & 31;
  int st   = tid >> 5;                 // s*nColTiles + tile
  int tile = st % nColTiles;
  int s    = st / nColTiles;
  int n     = tile * 16 + (lane & 15);
  int kbase = s * 32 + (lane >> 4) * 8;
  __bf16* dst = Wp + (size_t)tid * 16;
#pragma unroll
  for (int e = 0; e < 16; ++e) {
    float v = (n < Nvalid) ? W[(size_t)(kbase + (e < 8 ? e : e + 8)) * Nvalid + n] : 0.0f;
    dst[e] = f2bf(v);
  }
}

// ------------------------------------------------------------- WMMA GEMM
// One wave computes a 16 x (16*T) tile of out = A(f32) @ W(bf16 packed).
// A fragment layout (16-bit, 16x32): lanes 0-15 = rows, K 0-7 & 16-23;
// lanes 16-31 same rows, K 8-15 & 24-31 (ISA 7.12.2).
// Epilogue stages the C tile in LDS (per-wave private, padded row stride)
// and issues fully coalesced 128B row-major stores.
template <int T>
__global__ void gemm_wmma_bf16(const float* __restrict__ A, const __bf16* __restrict__ Wp,
                               float* __restrict__ out, int K, int nRowTiles,
                               int nColGroups, int nColTiles, int outCols) {
  constexpr int TW = T * 16;
  __shared__ float lds[8][16][TW + 4];     // +4 pad: conflict-free ds writes

  int wid  = threadIdx.x >> 5;
  int lane = threadIdx.x & 31;
  int flat = blockIdx.x * 8 + wid;                 // 256 threads = 8 waves
  if (flat >= nRowTiles * nColGroups) return;
  int rowTile  = flat / nColGroups;
  int colGroup = flat - rowTile * nColGroups;

  int halfw = lane >> 4;
  int m     = lane & 15;
  const float* arow = A + (size_t)(rowTile * 16 + m) * K;
  int h8 = halfw * 8;

  v8f acc[T];
#pragma unroll
  for (int t = 0; t < T; ++t) acc[t] = v8f{0, 0, 0, 0, 0, 0, 0, 0};

  int nSteps = K >> 5;
  for (int s = 0; s < nSteps; ++s) {
    int kk = s << 5;
    float4 p0 = *(const float4*)(arow + kk + h8);
    float4 p1 = *(const float4*)(arow + kk + h8 + 4);
    float4 q0 = *(const float4*)(arow + kk + 16 + h8);
    float4 q1 = *(const float4*)(arow + kk + 16 + h8 + 4);
    v16bf a;
    a[0]  = f2bf(p0.x); a[1]  = f2bf(p0.y); a[2]  = f2bf(p0.z); a[3]  = f2bf(p0.w);
    a[4]  = f2bf(p1.x); a[5]  = f2bf(p1.y); a[6]  = f2bf(p1.z); a[7]  = f2bf(p1.w);
    a[8]  = f2bf(q0.x); a[9]  = f2bf(q0.y); a[10] = f2bf(q0.z); a[11] = f2bf(q0.w);
    a[12] = f2bf(q1.x); a[13] = f2bf(q1.y); a[14] = f2bf(q1.z); a[15] = f2bf(q1.w);
#pragma unroll
    for (int t = 0; t < T; ++t) {
      int tile = colGroup * T + t;
      const v16bf b =
          *reinterpret_cast<const v16bf*>(Wp + ((size_t)(s * nColTiles + tile) * 32 + lane) * 16);
      acc[t] = __builtin_amdgcn_wmma_f32_16x16x32_bf16(
          false, a, false, b, (short)0, acc[t], false, false);
    }
  }

  // ---- epilogue ----
  // C/D layout: lane n = lane&15, VGPR v -> row = v + (lane>>4)*8.
  float (*tile)[TW + 4] = lds[wid];
#pragma unroll
  for (int t = 0; t < T; ++t)
#pragma unroll
    for (int v = 0; v < 8; ++v)
      tile[halfw * 8 + v][t * 16 + m] = acc[t][v];

  // coalesced read-back: 2 rows x 16 lanes of float4 per iteration
  int colBase = colGroup * TW;
#pragma unroll
  for (int i = 0; i < 8; ++i) {
    int r  = i * 2 + halfw;
    int c4 = m * 4;
    if (c4 < TW) {
      int col = colBase + c4;
      if (col + 4 <= outCols) {
        float4 v4 = *(const float4*)&tile[r][c4];
        *(float4*)&out[(size_t)(rowTile * 16 + r) * outCols + col] = v4;
      }
    }
  }
}

// --------------------------------------------------------- message passing
// agg[n] = b + dinv[n]^2 * h[n]   (self-loop term, no atomics needed)
__global__ void agg_init(const float* __restrict__ h, const float* __restrict__ bias,
                         const float* __restrict__ dinv, float* __restrict__ agg, int F4) {
  int t = blockIdx.x * blockDim.x + threadIdx.x;
  if (t >= NNODES * F4) return;
  int n = t / F4, f = t - n * F4;
  float d  = dinv[n];
  float sl = d * d;
  float4 hv = ((const float4*)h)[t];
  float4 bv = ((const float4*)bias)[f];
  float4 o;
  o.x = bv.x + sl * hv.x; o.y = bv.y + sl * hv.y;
  o.z = bv.z + sl * hv.z; o.w = bv.w + sl * hv.w;
  ((float4*)agg)[t] = o;
}

// agg[row[e]] += norm[e] * h[col[e]]   (L2-resident f32 atomics)
__global__ void agg_edges(const float* __restrict__ h, const int* __restrict__ row,
                          const int* __restrict__ col, const float* __restrict__ norm,
                          float* __restrict__ agg, int F4) {
  long t = (long)blockIdx.x * blockDim.x + threadIdx.x;
  long total = (long)NEDGES * F4;
  if (t >= total) return;
  int e = (int)(t / F4);
  int f = (int)(t - (long)e * F4);
  float w = norm[e];
  float4 hv = ((const float4*)(h + (size_t)col[e] * F4 * 4))[f];
  float* dst = agg + (size_t)row[e] * F4 * 4 + f * 4;
  unsafeAtomicAdd(dst + 0, w * hv.x);
  unsafeAtomicAdd(dst + 1, w * hv.y);
  unsafeAtomicAdd(dst + 2, w * hv.z);
  unsafeAtomicAdd(dst + 3, w * hv.w);
}

// ----------------------------------------------------------- batchnorm+relu
__global__ void bn_stats(const float* __restrict__ h, float* __restrict__ sums,
                         float* __restrict__ sumsq) {
  int c = threadIdx.x;                         // 256 = HIDD
  float s = 0.f, s2 = 0.f;
  for (int r = blockIdx.x; r < NNODES; r += gridDim.x) {
    float v = h[(size_t)r * HIDD + c];
    s += v; s2 += v * v;
  }
  unsafeAtomicAdd(&sums[c], s);
  unsafeAtomicAdd(&sumsq[c], s2);
}

__global__ void bn_finalize(const float* __restrict__ sums, const float* __restrict__ sumsq,
                            const float* __restrict__ gamma, const float* __restrict__ beta,
                            float* __restrict__ scale, float* __restrict__ shift) {
  int c = threadIdx.x;
  float mu  = sums[c] * (1.0f / NNODES);
  float var = sumsq[c] * (1.0f / NNODES) - mu * mu;
  float inv = rsqrtf(var + BN_EPS);
  float sc  = gamma[c] * inv;
  scale[c] = sc;
  shift[c] = beta[c] - mu * sc;
}

__global__ void bn_apply_relu(float* __restrict__ h, const float* __restrict__ scale,
                              const float* __restrict__ shift) {
  int t = blockIdx.x * blockDim.x + threadIdx.x;
  if (t >= NNODES * (HIDD / 4)) return;
  int c = (t % (HIDD / 4)) * 4;
  float4 v = ((float4*)h)[t];
  v.x = fmaxf(0.f, v.x * scale[c + 0] + shift[c + 0]);
  v.y = fmaxf(0.f, v.y * scale[c + 1] + shift[c + 1]);
  v.z = fmaxf(0.f, v.z * scale[c + 2] + shift[c + 2]);
  v.w = fmaxf(0.f, v.w * scale[c + 3] + shift[c + 3]);
  ((float4*)h)[t] = v;
}

// ------------------------------------------------------------- log_softmax
__global__ void log_softmax_rows(const float* __restrict__ in, float* __restrict__ out) {
  int warp = (blockIdx.x * blockDim.x + threadIdx.x) >> 5;
  int lane = threadIdx.x & 31;
  if (warp >= NNODES) return;
  const float* r = in + (size_t)warp * OUTD;
  float v0 = r[lane];                                   // lanes 0..31 -> f 0..31
  bool  has2 = (lane < OUTD - 32);                      // lanes 0..7 -> f 32..39
  float v1 = has2 ? r[32 + lane] : -INFINITY;
  float m = fmaxf(v0, v1);
  for (int off = 16; off; off >>= 1) m = fmaxf(m, __shfl_xor(m, off, 32));
  float s = __expf(v0 - m) + (has2 ? __expf(v1 - m) : 0.0f);
  for (int off = 16; off; off >>= 1) s += __shfl_xor(s, off, 32);
  float lse = m + __logf(s);
  out[(size_t)warp * OUTD + lane] = v0 - lse;
  if (has2) out[(size_t)warp * OUTD + 32 + lane] = v1 - lse;
}

// ---------------------------------------------------------------------------
extern "C" void kernel_launch(void* const* d_in, const int* in_sizes, int n_in,
                              void* d_out, int out_size, void* d_ws, size_t ws_size,
                              hipStream_t stream) {
  const float* x     = (const float*)d_in[0];
  const int*   ei    = (const int*)d_in[1];
  const int*   row   = ei;               // edge_index[0]
  const int*   col   = ei + NEDGES;      // edge_index[1]
  const float* W1    = (const float*)d_in[2];
  const float* b1    = (const float*)d_in[3];
  const float* W2    = (const float*)d_in[4];
  const float* b2    = (const float*)d_in[5];
  const float* W3    = (const float*)d_in[6];
  const float* b3    = (const float*)d_in[7];
  const float* gamma = (const float*)d_in[8];
  const float* beta  = (const float*)d_in[9];
  float* out = (float*)d_out;

  // ---- workspace carve (all offsets 256B aligned) ----
  char* ws = (char*)d_ws;
  size_t off = 0;
  auto carve = [&](size_t bytes) -> char* {
    char* p = ws + off;
    off += (bytes + 255) & ~(size_t)255;
    return p;
  };
  float*  bufT  = (float*)carve((size_t)NNODES * HIDD * 4);  // GEMM outputs
  float*  bufA  = (float*)carve((size_t)NNODES * HIDD * 4);  // aggregates / activations
  float*  dinv  = (float*)carve((size_t)NNODES * 4);
  float*  normv = (float*)carve((size_t)NEDGES * 4);
  float*  sums  = (float*)carve(HIDD * 4);
  float*  sumsq = (float*)carve(HIDD * 4);   // contiguous after sums (1KB aligned)
  float*  scale = (float*)carve(HIDD * 4);
  float*  shift = (float*)carve(HIDD * 4);
  __bf16* W1p = (__bf16*)carve((size_t)(IN_DIM / 32) * (HIDD / 16) * 32 * 16 * 2);
  __bf16* W2p = (__bf16*)carve((size_t)(HIDD / 32) * (HIDD / 16) * 32 * 16 * 2);
  __bf16* W3p = (__bf16*)carve((size_t)(HIDD / 32) * (OUTP / 16) * 32 * 16 * 2);

  const int B = 256;
  auto blocks = [](long n, int b) { return (int)((n + b - 1) / b); };

  // ---- pack weights into WMMA-fragment layout ----
  pack_weight_bf16<<<blocks((IN_DIM / 32) * (HIDD / 16) * 32, B), B, 0, stream>>>(
      W1, W1p, IN_DIM, HIDD, HIDD / 16);
  pack_weight_bf16<<<blocks((HIDD / 32) * (HIDD / 16) * 32, B), B, 0, stream>>>(
      W2, W2p, HIDD, HIDD, HIDD / 16);
  pack_weight_bf16<<<blocks((HIDD / 32) * (OUTP / 16) * 32, B), B, 0, stream>>>(
      W3, W3p, HIDD, OUTD, OUTP / 16);

  // ---- symmetric normalization: deg = 1 (self loop) + in-degree ----
  fill_f32<<<blocks(NNODES, B), B, 0, stream>>>(dinv, NNODES, 1.0f);
  degree_accum<<<blocks(NEDGES, B), B, 0, stream>>>(col, dinv);
  rsqrt_inplace<<<blocks(NNODES, B), B, 0, stream>>>(dinv, NNODES);
  edge_norm<<<blocks(NEDGES, B), B, 0, stream>>>(row, col, dinv, normv);

  const int rowTiles = NNODES / 16;  // 3125

  // ================= layer 1 =================
  gemm_wmma_bf16<4><<<blocks((long)rowTiles * 4, 8), B, 0, stream>>>(
      x, W1p, bufT, IN_DIM, rowTiles, /*nColGroups=*/4, /*nColTiles=*/16, HIDD);
  agg_init<<<blocks((long)NNODES * (HIDD / 4), B), B, 0, stream>>>(bufT, b1, dinv, bufA, HIDD / 4);
  agg_edges<<<blocks((long)NEDGES * (HIDD / 4), B), B, 0, stream>>>(
      bufT, row, col, normv, bufA, HIDD / 4);
  // BN + ReLU (in place on bufA)
  fill_f32<<<blocks(2 * HIDD, B), B, 0, stream>>>(sums, 2 * HIDD, 0.0f);
  bn_stats<<<512, HIDD, 0, stream>>>(bufA, sums, sumsq);
  bn_finalize<<<1, HIDD, 0, stream>>>(sums, sumsq, gamma, beta, scale, shift);
  bn_apply_relu<<<blocks((long)NNODES * (HIDD / 4), B), B, 0, stream>>>(bufA, scale, shift);

  // ================= layer 2 =================
  gemm_wmma_bf16<4><<<blocks((long)rowTiles * 4, 8), B, 0, stream>>>(
      bufA, W2p, bufT, HIDD, rowTiles, 4, 16, HIDD);
  agg_init<<<blocks((long)NNODES * (HIDD / 4), B), B, 0, stream>>>(bufT, b2, dinv, bufA, HIDD / 4);
  agg_edges<<<blocks((long)NEDGES * (HIDD / 4), B), B, 0, stream>>>(
      bufT, row, col, normv, bufA, HIDD / 4);
  fill_f32<<<blocks(2 * HIDD, B), B, 0, stream>>>(sums, 2 * HIDD, 0.0f);
  bn_stats<<<512, HIDD, 0, stream>>>(bufA, sums, sumsq);
  bn_finalize<<<1, HIDD, 0, stream>>>(sums, sumsq, gamma, beta, scale, shift);
  bn_apply_relu<<<blocks((long)NNODES * (HIDD / 4), B), B, 0, stream>>>(bufA, scale, shift);

  // ================= layer 3 =================
  gemm_wmma_bf16<3><<<blocks((long)rowTiles, 8), B, 0, stream>>>(
      bufA, W3p, bufT, HIDD, rowTiles, /*nColGroups=*/1, /*nColTiles=*/3, OUTD);
  agg_init<<<blocks((long)NNODES * (OUTD / 4), B), B, 0, stream>>>(bufT, b3, dinv, bufA, OUTD / 4);
  agg_edges<<<blocks((long)NEDGES * (OUTD / 4), B), B, 0, stream>>>(
      bufT, row, col, normv, bufA, OUTD / 4);

  // ================= log_softmax =================
  log_softmax_rows<<<blocks((long)NNODES * 32, B), B, 0, stream>>>(bufA, out);
}
